// MusicSinc1D_28157805593069
// MI455X (gfx1250) — matrix operand order repaired and checked by reading, not compiled, not run on hardware
//
#include <hip/hip_runtime.h>
#include <hip/hip_bf16.h>

// ---------------------------------------------------------------------------
// SincNet conv1d for MI455X (gfx1250): filters built on device, conv done as
// an implicit-im2col GEMM on v_wmma_f32_16x16x32_f16.
//   M = 256 filters, K = 2501 taps (padded to 2528), N = 4*32000 times.
// Each wave owns 64 filters x 16 times so one B (signal) fragment feeds
// four WMMAs, amortizing the per-lane sliding-window LDS gather.
// ---------------------------------------------------------------------------

typedef __attribute__((ext_vector_type(16))) _Float16 v16h;
typedef __attribute__((ext_vector_type(8)))  _Float16 v8h;
typedef __attribute__((ext_vector_type(8)))  float    v8f;

#define N_FILT    256
#define FILT_DIM  2501
#define KPAD      2528            // 79 chunks of 32
#define N_TIME    32000
#define N_BATCH   4
#define XPAD_LEN  34560           // 1250 zeros | 32000 | tail zeros
#define PAD_LO    1250
#define FS_       16000.0f

// ---------------------------------------------------------------------------
// Kernel 1: build the 256 sinc filters (band-pass, max-normalized, Hamming),
// store as f16 rows of length KPAD (zero tail). One block per filter.
// ---------------------------------------------------------------------------
__global__ void sinc_build_filters(const float* __restrict__ filt_b1,
                                   const float* __restrict__ filt_band,
                                   _Float16* __restrict__ wflt) {
  const int f   = blockIdx.x;
  const int tid = threadIdx.x;

  const float fb = fabsf(filt_b1[f])  + 50.0f / FS_;           // filt_beg
  const float fe = fb + fabsf(filt_band[f]) + 50.0f / FS_;     // filt_end

  __shared__ float red[256];

  // pass 1: local max of band_pass
  float lmax = -3.4e38f;
  for (int j = tid; j < FILT_DIM; j += 256) {
    const int i = j - (FILT_DIM - 1) / 2;          // -1250..1250
    float bp;
    if (i == 0) {
      bp = 2.0f * (fe - fb);
    } else {
      const float ai = fabsf((float)i);
      const float a1 = 2.0f * fb * ai;             // = 2*band_hz*t
      const float a2 = 2.0f * fe * ai;
      bp = 2.0f * fe * (sinf(a2) / a2) - 2.0f * fb * (sinf(a1) / a1);
    }
    lmax = fmaxf(lmax, bp);
  }
  red[tid] = lmax;
  __syncthreads();
  for (int s = 128; s > 0; s >>= 1) {
    if (tid < s) red[tid] = fmaxf(red[tid], red[tid + s]);
    __syncthreads();
  }
  const float inv_m = 1.0f / red[0];

  // pass 2: recompute, normalize, window, cast to f16
  for (int j = tid; j < FILT_DIM; j += 256) {
    const int i = j - (FILT_DIM - 1) / 2;
    float bp;
    if (i == 0) {
      bp = 2.0f * (fe - fb);
    } else {
      const float ai = fabsf((float)i);
      const float a1 = 2.0f * fb * ai;
      const float a2 = 2.0f * fe * ai;
      bp = 2.0f * fe * (sinf(a2) / a2) - 2.0f * fb * (sinf(a1) / a1);
    }
    const float win = 0.54f - 0.46f * cosf(6.283185307179586f * (float)j / (float)FILT_DIM);
    wflt[(size_t)f * KPAD + j] = (_Float16)(bp * inv_m * win);
  }
  for (int j = FILT_DIM + tid; j < KPAD; j += 256)
    wflt[(size_t)f * KPAD + j] = (_Float16)0.0f;
}

// ---------------------------------------------------------------------------
// Kernel 2: zero-pad + f16-cast the signal: xpad[b][t'] = x[b][t'-1250] or 0.
// ---------------------------------------------------------------------------
__global__ void sinc_pad_cast(const float* __restrict__ x,
                              _Float16* __restrict__ xpad) {
  const int g = blockIdx.x * blockDim.x + threadIdx.x;
  if (g >= N_BATCH * XPAD_LEN) return;
  const int b = g / XPAD_LEN;
  const int t = g - b * XPAD_LEN;
  const int s = t - PAD_LO;
  const float v = (s >= 0 && s < N_TIME) ? x[(size_t)b * N_TIME + s] : 0.0f;
  xpad[g] = (_Float16)v;
}

// ---------------------------------------------------------------------------
// Kernel 3: WMMA GEMM.  Block = 256 threads (8 waves) covers 128 time steps
// and 64 filters.  x window staged in LDS once (b128 loads), reused by all
// waves across all 79 K-chunks.  Each wave: 4 filter tiles x 1 time tile,
// so each B fragment (16 ds_load_u16) feeds 4 v_wmma ops.
//
//   A (16x32 f16): lane = M (filter), lanes>=16 take K-halves +8/+24.
//   B (32x16 f16): lane = N (time),   lanes>=16 take K-halves +8/+24.
//   D (16x16 f32): lane = N, VGPR r = M (lane>=16 -> M=r+8).
// ---------------------------------------------------------------------------
#define XS_ELEMS 2656   // 128 + 2528, multiple of 8

__device__ __forceinline__ v16h cat8(v8h lo, v8h hi) {
  return __builtin_shufflevector(lo, hi, 0, 1, 2, 3, 4, 5, 6, 7,
                                 8, 9, 10, 11, 12, 13, 14, 15);
}

__global__ void __launch_bounds__(256)
sinc_conv_wmma(const _Float16* __restrict__ wflt,
               const _Float16* __restrict__ xpad,
               float* __restrict__ out) {
  const int tchunk = blockIdx.x;            // 0..249  -> 128 times each
  const int fgrp   = blockIdx.y;            // 0..3    -> 64 filters each
  const int bz     = blockIdx.z;            // batch
  const int tbase  = tchunk * 128;
  const int fbase  = fgrp * 64;

  __shared__ _Float16 xs[XS_ELEMS];

  // stage x window into LDS with 16B vector copies (both sides 16B aligned)
  {
    const uint4* src = reinterpret_cast<const uint4*>(xpad + (size_t)bz * XPAD_LEN + tbase);
    uint4* dst = reinterpret_cast<uint4*>(xs);
    for (int i = threadIdx.x; i < XS_ELEMS / 8; i += 256) dst[i] = src[i];
  }
  __syncthreads();

  const int wave  = threadIdx.x >> 5;       // 0..7 -> time sub-tile
  const int lane  = threadIdx.x & 31;
  const int n     = lane & 15;              // N index (time within tile)
  const int hi    = lane >> 4;              // lane half
  const int koffA = hi * 8;                 // A K-offset for this half
  const int koffB = hi * 8;                 // B K-offset for this half
  const int twave = wave * 16;

  const _Float16* wrow0 = wflt + (size_t)(fbase + n) * KPAD;        // A lane row
  const _Float16* wrow1 = wrow0 + (size_t)16 * KPAD;
  const _Float16* wrow2 = wrow0 + (size_t)32 * KPAD;
  const _Float16* wrow3 = wrow0 + (size_t)48 * KPAD;
  const _Float16* xb    = xs + twave + n + koffB;                   // B lane column

  v8f c0 = {};
  v8f c1 = {};
  v8f c2 = {};
  v8f c3 = {};

  for (int kc = 0; kc < KPAD; kc += 32) {
    // B fragment: per-lane sliding window from LDS (2B granularity -> u16)
    v16h bf;
#pragma unroll
    for (int j = 0; j < 16; ++j) bf[j] = xb[kc + j + (j >= 8 ? 8 : 0)];

    // A fragments: two aligned 16B loads per tile (K-contiguous filter rows)
    const v16h a0 = cat8(*reinterpret_cast<const v8h*>(wrow0 + kc + koffA),
                         *reinterpret_cast<const v8h*>(wrow0 + kc + 16 + koffA));
    const v16h a1 = cat8(*reinterpret_cast<const v8h*>(wrow1 + kc + koffA),
                         *reinterpret_cast<const v8h*>(wrow1 + kc + 16 + koffA));
    const v16h a2 = cat8(*reinterpret_cast<const v8h*>(wrow2 + kc + koffA),
                         *reinterpret_cast<const v8h*>(wrow2 + kc + 16 + koffA));
    const v16h a3 = cat8(*reinterpret_cast<const v8h*>(wrow3 + kc + koffA),
                         *reinterpret_cast<const v8h*>(wrow3 + kc + 16 + koffA));
    if (kc + 32 < KPAD) {                    // stream next filter chunks
      __builtin_prefetch(wrow0 + kc + 32 + koffA, 0, 0);
      __builtin_prefetch(wrow1 + kc + 32 + koffA, 0, 0);
      __builtin_prefetch(wrow2 + kc + 32 + koffA, 0, 0);
      __builtin_prefetch(wrow3 + kc + 32 + koffA, 0, 0);
    }

    c0 = __builtin_amdgcn_wmma_f32_16x16x32_f16(false, a0, false, bf,
                                                (short)0, c0, false, false);
    c1 = __builtin_amdgcn_wmma_f32_16x16x32_f16(false, a1, false, bf,
                                                (short)0, c1, false, false);
    c2 = __builtin_amdgcn_wmma_f32_16x16x32_f16(false, a2, false, bf,
                                                (short)0, c2, false, false);
    c3 = __builtin_amdgcn_wmma_f32_16x16x32_f16(false, a3, false, bf,
                                                (short)0, c3, false, false);
  }

  // store: lane owns time t, 8 consecutive filters per tile (32B aligned)
  const int t = tbase + twave + n;
  float* o = out + (((size_t)bz * N_TIME + t) * N_FILT) + fbase + hi * 8;
  float4 s0 = {c0[0], c0[1], c0[2], c0[3]};
  float4 s1 = {c0[4], c0[5], c0[6], c0[7]};
  float4 s2 = {c1[0], c1[1], c1[2], c1[3]};
  float4 s3 = {c1[4], c1[5], c1[6], c1[7]};
  float4 s4 = {c2[0], c2[1], c2[2], c2[3]};
  float4 s5 = {c2[4], c2[5], c2[6], c2[7]};
  float4 s6 = {c3[0], c3[1], c3[2], c3[3]};
  float4 s7 = {c3[4], c3[5], c3[6], c3[7]};
  *reinterpret_cast<float4*>(o)      = s0;
  *reinterpret_cast<float4*>(o + 4)  = s1;
  *reinterpret_cast<float4*>(o + 16) = s2;
  *reinterpret_cast<float4*>(o + 20) = s3;
  *reinterpret_cast<float4*>(o + 32) = s4;
  *reinterpret_cast<float4*>(o + 36) = s5;
  *reinterpret_cast<float4*>(o + 48) = s6;
  *reinterpret_cast<float4*>(o + 52) = s7;
}

// ---------------------------------------------------------------------------
extern "C" void kernel_launch(void* const* d_in, const int* in_sizes, int n_in,
                              void* d_out, int out_size, void* d_ws, size_t ws_size,
                              hipStream_t stream) {
  const float* x         = (const float*)d_in[0];   // (4, 32000, 1)
  const float* filt_b1   = (const float*)d_in[1];   // (256, 1)
  const float* filt_band = (const float*)d_in[2];   // (256, 1)
  float* out = (float*)d_out;                       // (4, 32000, 256)

  // workspace layout: f16 filters (256 x 2528) then f16 padded signal (4 x 34560)
  _Float16* wflt = (_Float16*)d_ws;
  _Float16* xpad = (_Float16*)((char*)d_ws + (size_t)N_FILT * KPAD * sizeof(_Float16));

  sinc_build_filters<<<N_FILT, 256, 0, stream>>>(filt_b1, filt_band, wflt);

  const int pad_total = N_BATCH * XPAD_LEN;
  sinc_pad_cast<<<(pad_total + 255) / 256, 256, 0, stream>>>(x, xpad);

  dim3 grid(N_TIME / 128, N_FILT / 64, N_BATCH);   // (250, 4, 4)
  sinc_conv_wmma<<<grid, 256, 0, stream>>>(wflt, xpad, out);
}
